// LinearBit_81758997446675
// MI455X (gfx1250) — matrix operand order repaired and checked by reading, not compiled, not run on hardware
//
#include <hip/hip_runtime.h>

// ---------------------------------------------------------------------------
// out[8192,4096] = x[8192,4096] @ (alpha * w)[4096,4096]
//   w = sign(sign_bit) * (bit0*0.5 + bit1*0.25 + bit2*0.125), bit_i = (mag_i>=0)
//   alpha = sqrt(2/4096) / (std(w) + 1e-12)
// Pipeline: fused build+reduce -> alpha -> WMMA GEMM (f16 in, f32 accum).
// GEMM: 256x128 block tile, 64x64 wave tile (16 v_wmma per K-step), double-
// buffered LDS, B tile streamed with global_load_async_to_lds_b128 (ASYNCcnt).
// ---------------------------------------------------------------------------

typedef _Float16 v16h __attribute__((ext_vector_type(16)));
typedef _Float16 v8h  __attribute__((ext_vector_type(8)));
typedef float    v8f  __attribute__((ext_vector_type(8)));

#define M_DIM 8192
#define K_DIM 4096
#define N_DIM 4096
#define N_ELEM (K_DIM * N_DIM)

#define TM 256
#define TN 128
#define TK 32
#define LDK 40                     // padded LDS row (halfwords)
#define KTILES (K_DIM / TK)

// ---------------------------------------------------------------------------
__global__ __launch_bounds__(32) void bw_init_kernel(float* __restrict__ sums) {
    if (threadIdx.x == 0) { sums[0] = 0.0f; sums[1] = 0.0f; }
}

// ---------------------------------------------------------------------------
// Fused weight build (f16, transposed) + sum / sum-of-squares reduction.
__global__ __launch_bounds__(256) void bw_build_kernel(
    const float* __restrict__ mag, const float* __restrict__ sgn,
    _Float16* __restrict__ wT, float* __restrict__ sums)
{
    __shared__ _Float16 tile[32][33];
    __shared__ float bs[2];
    const int tx = threadIdx.x & 31;
    const int ty = threadIdx.x >> 5;
    if (threadIdx.x == 0) { bs[0] = 0.0f; bs[1] = 0.0f; }

    const int nBase = blockIdx.x * 32;
    const int kBase = blockIdx.y * 32;
    float s1 = 0.0f, s2 = 0.0f;
#pragma unroll
    for (int r = 0; r < 4; ++r) {
        const int k = kBase + ty + r * 8;
        const int n = nBase + tx;
        const size_t idx = (size_t)k * N_DIM + n;
        float m = 0.0f;
        m += (mag[idx]                      >= 0.0f) ? 0.500f : 0.0f;
        m += (mag[idx + (size_t)N_ELEM]     >= 0.0f) ? 0.250f : 0.0f;
        m += (mag[idx + 2 * (size_t)N_ELEM] >= 0.0f) ? 0.125f : 0.0f;
        const float w = (sgn[idx] >= 0.0f) ? m : -m;
        s1 += w;
        s2 += m * m;
        tile[ty + r * 8][tx] = (_Float16)w;
    }
    __syncthreads();
#pragma unroll
    for (int r = 0; r < 4; ++r) {
        const int n = nBase + ty + r * 8;
        const int k = kBase + tx;
        wT[(size_t)n * K_DIM + k] = tile[tx][ty + r * 8];
    }
    atomicAdd(&bs[0], s1);
    atomicAdd(&bs[1], s2);
    __syncthreads();
    if (threadIdx.x == 0) {
        atomicAdd(&sums[0], bs[0]);
        atomicAdd(&sums[1], bs[1]);
    }
}

// ---------------------------------------------------------------------------
__global__ __launch_bounds__(32) void bw_alpha_kernel(
    const float* __restrict__ sums, float* __restrict__ alpha)
{
    if (threadIdx.x == 0) {
        const float inv_n = 1.0f / (float)N_ELEM;
        const float mean  = sums[0] * inv_n;
        float var = sums[1] * inv_n - mean * mean;
        var = var > 0.0f ? var : 0.0f;
        const float stdv = sqrtf(var);
        *alpha = 0.02209708691207961f / (stdv + 1e-12f);   // sqrt(2/4096)/std
    }
}

// ---------------------------------------------------------------------------
// WMMA GEMM. Block: 256 threads = 8 wave32 (4 along M x 2 along N).
// Wave tile 64x64 = 4x4 fragments -> 16 v_wmma_f32_16x16x32_f16 per K-tile.
// Double-buffered LDS; B streamed via async load-to-LDS, A staged through
// VGPRs (f32->f16 convert) and stored after the WMMA block to hide latency.
// ---------------------------------------------------------------------------
__global__ __launch_bounds__(256) void bw_gemm_kernel(
    const float* __restrict__ x, const _Float16* __restrict__ wT,
    const float* __restrict__ alpha_p, float* __restrict__ out)
{
    __shared__ _Float16 lds_a[2][TM * LDK];   // 2 x 20 KiB
    __shared__ _Float16 lds_b[2][TN * LDK];   // 2 x 10 KiB

    const int tid   = threadIdx.x;
    const int lane  = tid & 31;
    const int wave  = tid >> 5;
    const int waveM = wave >> 1;           // 0..3
    const int waveN = wave & 1;            // 0..1
    const int m0 = blockIdx.y * TM;
    const int n0 = blockIdx.x * TN;
    const int lrow = lane & 15;            // fragment row/col within 16
    const int kh   = lane >> 4;            // K-half select (ISA 7.12.2)

    v8f acc[4][4] = {};

    // A loader: 256x32 f32 tile; 4 threads/row, 4 row passes (rows +64).
    const int aRow = tid >> 2;             // 0..63
    const int aCol = (tid & 3) * 8;        // float == halfword column
    const float* aPtr = x + (size_t)(m0 + aRow) * K_DIM + aCol;

    // B loader: 128x32 f16 tile; one 16B async chunk per (row, colgroup),
    // 2 passes (rows +64).
    const int bRow = tid >> 2;             // 0..63
    const int bCol = (tid & 3) * 8;        // halfword column
    const uint64_t bAddr0 =
        (uint64_t)(uintptr_t)(wT + (size_t)(n0 + bRow) * K_DIM + bCol);
    const uint64_t bAddr1 = bAddr0 + (uint64_t)64 * K_DIM * sizeof(_Float16);
    const unsigned bLds0 =
        (unsigned)(uintptr_t)&lds_b[0][bRow * LDK + bCol];
    const unsigned bLds1 =
        (unsigned)(uintptr_t)&lds_b[0][(bRow + 64) * LDK + bCol];
    const unsigned bufStrideB = (unsigned)(TN * LDK * sizeof(_Float16));

    // ---- prologue: stage K-tile 0 into buffer 0
    asm volatile("global_load_async_to_lds_b128 %0, %1, off"
                 :: "v"(bLds0), "v"(bAddr0) : "memory");
    asm volatile("global_load_async_to_lds_b128 %0, %1, off"
                 :: "v"(bLds1), "v"(bAddr1) : "memory");
#pragma unroll
    for (int r = 0; r < 4; ++r) {
        const float4* s = (const float4*)(aPtr + (size_t)(64 * r) * K_DIM);
        const float4 f0 = s[0], f1 = s[1];
        const v8h h = { (_Float16)f0.x, (_Float16)f0.y, (_Float16)f0.z, (_Float16)f0.w,
                        (_Float16)f1.x, (_Float16)f1.y, (_Float16)f1.z, (_Float16)f1.w };
        *(v8h*)&lds_a[0][(aRow + 64 * r) * LDK + aCol] = h;
    }
    asm volatile("s_wait_asynccnt 0" ::: "memory");
    __syncthreads();

    for (int kt = 0; kt < KTILES; ++kt) {
        const int cur = kt & 1;
        const int nxt = cur ^ 1;
        const bool hasNext = (kt + 1) < KTILES;

        float4 st[8];
        if (hasNext) {
            const int kOff = (kt + 1) * TK;                 // halfwords/floats
            // async B copy for next tile (overlaps the WMMA block below)
            const uint64_t ga0 = bAddr0 + (uint64_t)kOff * sizeof(_Float16);
            const uint64_t ga1 = bAddr1 + (uint64_t)kOff * sizeof(_Float16);
            const unsigned ld0 = bLds0 + (unsigned)nxt * bufStrideB;
            const unsigned ld1 = bLds1 + (unsigned)nxt * bufStrideB;
            asm volatile("global_load_async_to_lds_b128 %0, %1, off"
                         :: "v"(ld0), "v"(ga0) : "memory");
            asm volatile("global_load_async_to_lds_b128 %0, %1, off"
                         :: "v"(ld1), "v"(ga1) : "memory");
            // A global loads for next tile into registers
#pragma unroll
            for (int r = 0; r < 4; ++r) {
                const float4* s =
                    (const float4*)(aPtr + (size_t)(64 * r) * K_DIM + kOff);
                st[2 * r]     = s[0];
                st[2 * r + 1] = s[1];
            }
        }

        // ---- 16 WMMAs on current buffers
        union AF { v16h v; v8h h[2]; };
        AF af[4];
        const _Float16* la = lds_a[cur];
        const _Float16* lb = lds_b[cur];
#pragma unroll
        for (int fm = 0; fm < 4; ++fm) {
            const int ar = (waveM * 64 + fm * 16 + lrow) * LDK;
            af[fm].h[0] = *(const v8h*)&la[ar + kh * 8];
            af[fm].h[1] = *(const v8h*)&la[ar + 16 + kh * 8];
        }
#pragma unroll
        for (int fn = 0; fn < 4; ++fn) {
            const int br = (waveN * 64 + fn * 16 + lrow) * LDK;
            AF bu;
            bu.h[0] = *(const v8h*)&lb[br + kh * 8];
            bu.h[1] = *(const v8h*)&lb[br + 16 + kh * 8];
#pragma unroll
            for (int fm = 0; fm < 4; ++fm) {
                acc[fm][fn] = __builtin_amdgcn_wmma_f32_16x16x32_f16(
                    false, af[fm].v, false, bu.v,
                    (short)0, acc[fm][fn], false, false);
            }
        }

        if (hasNext) {
            // convert + stage next A tile into the other buffer
#pragma unroll
            for (int r = 0; r < 4; ++r) {
                const float4 f0 = st[2 * r], f1 = st[2 * r + 1];
                const v8h h = { (_Float16)f0.x, (_Float16)f0.y, (_Float16)f0.z, (_Float16)f0.w,
                                (_Float16)f1.x, (_Float16)f1.y, (_Float16)f1.z, (_Float16)f1.w };
                *(v8h*)&lds_a[nxt][(aRow + 64 * r) * LDK + aCol] = h;
            }
            asm volatile("s_wait_asynccnt 0" ::: "memory");
        }
        __syncthreads();
    }

    // ---- epilogue: scale by alpha, f32 stores (C/D layout per ISA 7.12.2)
    const float alpha = *alpha_p;
    const int colBase = n0 + waveN * 64 + lrow;
#pragma unroll
    for (int fm = 0; fm < 4; ++fm) {
        const int rowBase = m0 + waveM * 64 + fm * 16 + kh * 8;
#pragma unroll
        for (int fn = 0; fn < 4; ++fn) {
            const int col = colBase + fn * 16;
#pragma unroll
            for (int j = 0; j < 8; ++j) {
                out[(size_t)(rowBase + j) * N_DIM + col] = acc[fm][fn][j] * alpha;
            }
        }
    }
}

// ---------------------------------------------------------------------------
extern "C" void kernel_launch(void* const* d_in, const int* in_sizes, int n_in,
                              void* d_out, int out_size, void* d_ws, size_t ws_size,
                              hipStream_t stream) {
    const float* x   = (const float*)d_in[0];   // [8192, 4096]
    const float* mag = (const float*)d_in[1];   // [3, 4096, 4096]
    const float* sgn = (const float*)d_in[2];   // [4096, 4096]
    float* out = (float*)d_out;                 // [8192, 4096]

    float*    sums  = (float*)d_ws;                     // [0]=sum(w), [1]=sum(w^2)
    float*    alpha = sums + 2;                         // [2]=alpha
    _Float16* wT    = (_Float16*)((char*)d_ws + 256);   // 4096x4096 f16 (32 MB)

    bw_init_kernel<<<dim3(1), dim3(32), 0, stream>>>(sums);

    dim3 bgrid(N_DIM / 32, K_DIM / 32);
    bw_build_kernel<<<bgrid, dim3(256), 0, stream>>>(mag, sgn, wT, sums);

    bw_alpha_kernel<<<dim3(1), dim3(32), 0, stream>>>(sums, alpha);

    dim3 ggrid(N_DIM / TN, M_DIM / TM);
    bw_gemm_kernel<<<ggrid, dim3(256), 0, stream>>>(x, wT, alpha, out);
}
